// FullHierarchicalBEM_57226144252156
// MI455X (gfx1250) — compile-verified, matrix operand order checked by reading, not run on hardware
//
#include <hip/hip_runtime.h>
#include <hip/hip_bf16.h>

// ---------------------------------------------------------------------------
// Problem constants (from reference): B=4, S=4096, DIN=DOUT=4096, RANK=8
// ---------------------------------------------------------------------------
#define M_TOT 16384   // B*S token rows
#define K_TOT 4096    // DIN
#define N_TOT 4096    // DOUT
#define RANK  8
#define SCALING 2.0f

typedef __attribute__((ext_vector_type(16))) __bf16 v16bf;
typedef __attribute__((ext_vector_type(8)))  float  v8f;

union Frag16 { v16bf v; uint4 q[2]; };
union Pack8  { unsigned short s[8]; uint4 q; };

__device__ __forceinline__ unsigned short f2bf(float f) {
    union { float f; unsigned u; } c; c.f = f;
    unsigned u = c.u;
    unsigned r = u + 0x7FFFu + ((u >> 16) & 1u);   // RNE fp32 -> bf16
    return (unsigned short)(r >> 16);
}

#if __has_builtin(__builtin_amdgcn_s_wait_asynccnt)
#define WAIT_ASYNC(n) __builtin_amdgcn_s_wait_asynccnt(n)
#else
#define WAIT_ASYNC(n) asm volatile("s_wait_asynccnt %0" :: "i"(n))
#endif

// CDNA5 async global->LDS copy (16B), tracked by ASYNCcnt. No VGPR staging.
__device__ __forceinline__ void async_g2l_b128(unsigned lds_off,
                                               const unsigned short* g) {
    unsigned long long ga = (unsigned long long)(const void*)g;
    asm volatile("global_load_async_to_lds_b128 %0, %1, off"
                 :: "v"(lds_off), "v"(ga) : "memory");
}

// ---------------------------------------------------------------------------
// Kernel 1: per token row -- cast X row to bf16, compute
//   xvc[m,r] = (sum_i x[m,i] * V[i,r]) * codes[m,r] * SCALING
// ---------------------------------------------------------------------------
__global__ __launch_bounds__(256)
void prep_x_kernel(const float* __restrict__ x,
                   const float* __restrict__ codes,
                   const float* __restrict__ V,       // [K_TOT][RANK]
                   unsigned short* __restrict__ Xb,   // [M_TOT][K_TOT] bf16
                   float* __restrict__ xvc)           // [M_TOT][RANK]
{
    const int row = blockIdx.x;
    const int tid = threadIdx.x;
    const int lane = tid & 31;
    const int wave = tid >> 5;

    __shared__ float sRed[8 * RANK];

    const float* xr = x + (size_t)row * K_TOT;
    unsigned short* xb = Xb + (size_t)row * K_TOT;

    float acc[RANK];
#pragma unroll
    for (int r = 0; r < RANK; ++r) acc[r] = 0.0f;

    for (int i = tid; i < K_TOT; i += 256) {
        float xv = xr[i];
        xb[i] = f2bf(xv);
        const float4 v0 = *(const float4*)(V + (size_t)i * RANK);
        const float4 v1 = *(const float4*)(V + (size_t)i * RANK + 4);
        acc[0] += xv * v0.x; acc[1] += xv * v0.y;
        acc[2] += xv * v0.z; acc[3] += xv * v0.w;
        acc[4] += xv * v1.x; acc[5] += xv * v1.y;
        acc[6] += xv * v1.z; acc[7] += xv * v1.w;
    }

#pragma unroll
    for (int r = 0; r < RANK; ++r) {
#pragma unroll
        for (int off = 16; off > 0; off >>= 1)
            acc[r] += __shfl_down(acc[r], off, 32);
    }
    if (lane == 0) {
#pragma unroll
        for (int r = 0; r < RANK; ++r) sRed[wave * RANK + r] = acc[r];
    }
    __syncthreads();
    if (tid < RANK) {
        float t = 0.0f;
#pragma unroll
        for (int w = 0; w < 8; ++w) t += sRed[w * RANK + tid];
        xvc[(size_t)row * RANK + tid] =
            t * codes[(size_t)row * RANK + tid] * SCALING;
    }
}

// ---------------------------------------------------------------------------
// Kernel 2: cast W [N_TOT][K_TOT] fp32 -> bf16
// ---------------------------------------------------------------------------
__global__ __launch_bounds__(256)
void cast_w_kernel(const float* __restrict__ W, unsigned short* __restrict__ Wb)
{
    size_t i8 = ((size_t)blockIdx.x * 256 + threadIdx.x) * 8;
    float4 a = *(const float4*)(W + i8);
    float4 b = *(const float4*)(W + i8 + 4);
    Pack8 p;
    p.s[0] = f2bf(a.x); p.s[1] = f2bf(a.y); p.s[2] = f2bf(a.z); p.s[3] = f2bf(a.w);
    p.s[4] = f2bf(b.x); p.s[5] = f2bf(b.y); p.s[6] = f2bf(b.z); p.s[7] = f2bf(b.w);
    *(uint4*)(Wb + i8) = p.q;
}

// ---------------------------------------------------------------------------
// Kernel 3: fused GEMM + bias + rank-8 LoRA epilogue, double-buffered LDS
// with async global->LDS copies (ASYNCcnt pipeline).
//   out[m,n] = sum_k Xb[m,k]*Wb[n,k] + bias[n] + sum_r xvc[m,r]*U[n,r]
// Block tile 128x128, 256 threads = 8 wave32; each wave: 32x64 = 2x4 tiles of
// v_wmma_f32_16x16x32_bf16 (fp32 accumulation).
// ---------------------------------------------------------------------------
#define BT    128          // block tile (M and N)
#define KT    32           // K slab per LDS stage (one WMMA K-step)
#define LDA2  40           // padded LDS row stride in halfs (bank-disjoint)

__global__ __launch_bounds__(256)
void gemm_lora_kernel(const unsigned short* __restrict__ Xb,  // [M_TOT][K_TOT]
                      const unsigned short* __restrict__ Wb,  // [N_TOT][K_TOT]
                      const float* __restrict__ bias,         // [N_TOT]
                      const float* __restrict__ U,            // [N_TOT][RANK]
                      const float* __restrict__ xvc,          // [M_TOT][RANK]
                      float* __restrict__ out)                // [M_TOT][N_TOT]
{
    __shared__ unsigned short sX[2][BT * LDA2];   // 2 x 10240 B
    __shared__ unsigned short sW[2][BT * LDA2];   // 2 x 10240 B
    __shared__ float sU[BT * RANK];               // 4096 B
    __shared__ float sXvc[BT * RANK];             // 4096 B
    __shared__ float sBias[BT];                   // 512 B

    const int tid    = threadIdx.x;
    const int lane   = tid & 31;
    const int laneLo = lane & 15;
    const int laneHi = lane >> 4;                 // 0 or 1
    const int wave   = tid >> 5;
    const int waveM  = wave >> 1;                 // 0..3 -> row offset *32
    const int waveN  = wave & 1;                  // 0..1 -> col offset *64

    const int tileN = blockIdx.x * BT;
    const int tileM = blockIdx.y * BT;

    // ---- stage epilogue operands (bias, U rows, xvc rows) ----
    {
        int r = tid >> 1, h = tid & 1;
        *(float4*)&sU[r * RANK + h * 4] =
            *(const float4*)(U + (size_t)(tileN + r) * RANK + h * 4);
        *(float4*)&sXvc[r * RANK + h * 4] =
            *(const float4*)(xvc + (size_t)(tileM + r) * RANK + h * 4);
        if (tid < BT) sBias[tid] = bias[tileN + tid];
    }

    // ---- per-thread staging coordinates: 2 x16B chunks per matrix per slab
    // slab = 128 rows x 32 halfs = 512 x 16B chunks; 256 threads -> 2 each.
    const int v0 = tid, v1 = tid + 256;
    const int r0 = v0 >> 2, c0 = (v0 & 3) * 8;    // 4 chunks per row
    const int r1 = v1 >> 2, c1 = (v1 & 3) * 8;

    unsigned sx0[2], sx1[2], sw0[2], sw1[2];
#pragma unroll
    for (int bufi = 0; bufi < 2; ++bufi) {
        sx0[bufi] = (unsigned)(size_t)(const void*)&sX[bufi][r0 * LDA2 + c0];
        sx1[bufi] = (unsigned)(size_t)(const void*)&sX[bufi][r1 * LDA2 + c1];
        sw0[bufi] = (unsigned)(size_t)(const void*)&sW[bufi][r0 * LDA2 + c0];
        sw1[bufi] = (unsigned)(size_t)(const void*)&sW[bufi][r1 * LDA2 + c1];
    }
    const unsigned short* gx0 = Xb + (size_t)(tileM + r0) * K_TOT + c0;
    const unsigned short* gx1 = Xb + (size_t)(tileM + r1) * K_TOT + c1;
    const unsigned short* gw0 = Wb + (size_t)(tileN + r0) * K_TOT + c0;
    const unsigned short* gw1 = Wb + (size_t)(tileN + r1) * K_TOT + c1;

    v8f acc[2][4];
#pragma unroll
    for (int i = 0; i < 2; ++i)
#pragma unroll
        for (int j = 0; j < 4; ++j) acc[i][j] = (v8f)(0.0f);

    // ---- prologue: async-stage slab 0 into buffer 0 ----
    async_g2l_b128(sx0[0], gx0);
    async_g2l_b128(sx1[0], gx1);
    async_g2l_b128(sw0[0], gw0);
    async_g2l_b128(sw1[0], gw1);

    int buf = 0;
    for (int kt = 0; kt < K_TOT; kt += KT) {
        const bool more = (kt + KT) < K_TOT;
        if (more) {                      // prefetch next slab into other buffer
            async_g2l_b128(sx0[buf ^ 1], gx0 + kt + KT);
            async_g2l_b128(sx1[buf ^ 1], gx1 + kt + KT);
            async_g2l_b128(sw0[buf ^ 1], gw0 + kt + KT);
            async_g2l_b128(sw1[buf ^ 1], gw1 + kt + KT);
            WAIT_ASYNC(4);               // in-order: oldest 4 (= this buf) done
        } else {
            WAIT_ASYNC(0);
        }
        __syncthreads();                 // all waves' async copies visible

        // ---- one WMMA K-step of 32 on buffer `buf` ----
        Frag16 a[2], b[4];
#pragma unroll
        for (int sm = 0; sm < 2; ++sm) {
            int r = waveM * 32 + sm * 16 + laneLo;
            int base = r * LDA2 + laneHi * 8;
            a[sm].q[0] = *(const uint4*)&sX[buf][base];
            a[sm].q[1] = *(const uint4*)&sX[buf][base + 16];
        }
#pragma unroll
        for (int sn = 0; sn < 4; ++sn) {
            int c = waveN * 64 + sn * 16 + laneLo;
            int base = c * LDA2 + laneHi * 8;
            b[sn].q[0] = *(const uint4*)&sW[buf][base];
            b[sn].q[1] = *(const uint4*)&sW[buf][base + 16];
        }
#pragma unroll
        for (int sm = 0; sm < 2; ++sm)
#pragma unroll
            for (int sn = 0; sn < 4; ++sn)
                acc[sm][sn] = __builtin_amdgcn_wmma_f32_16x16x32_bf16(
                    false, a[sm].v, false, b[sn].v,
                    (short)0, acc[sm][sn], false, false);

        __syncthreads();                 // all waves done reading before reuse
        buf ^= 1;
    }

    // ---- epilogue: bias + rank-8 LoRA correction, fp32 ----
#pragma unroll
    for (int sn = 0; sn < 4; ++sn) {
        int n_loc = waveN * 64 + sn * 16 + laneLo;
        float bn = sBias[n_loc];
        float4 u0 = *(const float4*)&sU[n_loc * RANK];
        float4 u1 = *(const float4*)&sU[n_loc * RANK + 4];
        size_t gn = (size_t)tileN + n_loc;
#pragma unroll
        for (int sm = 0; sm < 2; ++sm) {
#pragma unroll
            for (int j = 0; j < 8; ++j) {
                int m_loc = waveM * 32 + sm * 16 + laneHi * 8 + j;
                float4 x0 = *(const float4*)&sXvc[m_loc * RANK];
                float4 x1 = *(const float4*)&sXvc[m_loc * RANK + 4];
                float lora = u0.x * x0.x + u0.y * x0.y + u0.z * x0.z + u0.w * x0.w
                           + u1.x * x1.x + u1.y * x1.y + u1.z * x1.z + u1.w * x1.w;
                out[(size_t)(tileM + m_loc) * N_TOT + gn] =
                    acc[sm][sn][j] + bn + lora;
            }
        }
    }
}

// ---------------------------------------------------------------------------
// Host launcher. d_in order: x, codes, W, b, lora_U, lora_V (all fp32).
// Workspace: Xb bf16 (128 MB) | Wb bf16 (32 MB) | xvc fp32 (0.5 MB)
// ---------------------------------------------------------------------------
extern "C" void kernel_launch(void* const* d_in, const int* in_sizes, int n_in,
                              void* d_out, int out_size, void* d_ws, size_t ws_size,
                              hipStream_t stream) {
    const float* x     = (const float*)d_in[0];
    const float* codes = (const float*)d_in[1];
    const float* W     = (const float*)d_in[2];
    const float* b     = (const float*)d_in[3];
    const float* U     = (const float*)d_in[4];
    const float* V     = (const float*)d_in[5];
    float* out = (float*)d_out;

    char* ws = (char*)d_ws;
    unsigned short* Xb = (unsigned short*)ws;
    unsigned short* Wb = (unsigned short*)(ws + (size_t)M_TOT * K_TOT * 2);
    float*          xvc = (float*)(ws + (size_t)M_TOT * K_TOT * 2
                                      + (size_t)N_TOT * K_TOT * 2);

    prep_x_kernel<<<M_TOT, 256, 0, stream>>>(x, codes, V, Xb, xvc);
    cast_w_kernel<<<(N_TOT * (size_t)K_TOT) / (256 * 8), 256, 0, stream>>>(W, Wb);

    dim3 grid(N_TOT / BT, M_TOT / BT);   // (32, 128)
    gemm_lora_kernel<<<grid, 256, 0, stream>>>(Xb, Wb, b, U, xvc, out);
}
